// ScaledDotProductAttentions_32263794328206
// MI455X (gfx1250) — matrix operand order repaired and checked by reading, not compiled, not run on hardware
//
#include <hip/hip_runtime.h>
#include <hip/hip_bf16.h>

typedef __attribute__((ext_vector_type(16))) __bf16 v16bf;
typedef __attribute__((ext_vector_type(8)))  __bf16 v8bf;
typedef __attribute__((ext_vector_type(4)))  __bf16 v4bf;
typedef __attribute__((ext_vector_type(8)))  float  v8f;
typedef __attribute__((ext_vector_type(4)))  float  v4f;

#define D_H      64
#define SEQ      2048
#define N_HEADS  16
#define N_BATCH  4
#define BR       64          // query rows per workgroup
#define BC       64          // key/value rows per iteration
#define LDSTR    72          // padded LDS row stride (bf16 elems), 144B (16B-aligned rows)
#define NKB      (SEQ / BC)  // 32 KV blocks
#define ATTN_SCALE 0.125f    // d_h^-0.5
#define LOG2E    1.44269504088896340736f

union FragA { v16bf v; v8bf h[2]; };

// ---- DPP16 butterfly reductions across a 16-lane half (wave32, pure VALU) ----
// Stages: quad_perm xor1 (0xB1), quad_perm xor2 (0x4E),
//         row_half_mirror (0x141) == xor4 at quad granularity,
//         row_mirror      (0x140) == xor8 at oct granularity.
template <int CTRL>
__device__ __forceinline__ float dpp_bfly(float x) {
    int y = __builtin_amdgcn_update_dpp(0, __builtin_bit_cast(int, x),
                                        CTRL, 0xF, 0xF, true);
    return __builtin_bit_cast(float, y);
}
__device__ __forceinline__ float row16_max(float x) {
    x = fmaxf(x, dpp_bfly<0x0B1>(x));
    x = fmaxf(x, dpp_bfly<0x04E>(x));
    x = fmaxf(x, dpp_bfly<0x141>(x));
    x = fmaxf(x, dpp_bfly<0x140>(x));
    return x;
}
__device__ __forceinline__ float row16_sum(float x) {
    x += dpp_bfly<0x0B1>(x);
    x += dpp_bfly<0x04E>(x);
    x += dpp_bfly<0x141>(x);
    x += dpp_bfly<0x140>(x);
    return x;
}

__global__ __launch_bounds__(128) void fa_fwd_bf16_wmma(
    const float* __restrict__ Q,
    const float* __restrict__ K,
    const float* __restrict__ V,
    float* __restrict__ O)
{
    __shared__ __bf16 Qs[BR * LDSTR];   // [qrow][d]
    __shared__ __bf16 Ks[BC * LDSTR];   // [krow][d]
    __shared__ __bf16 Vts[D_H * LDSTR]; // [d][krow]  (transposed V)
    __shared__ __bf16 Ps[BR * LDSTR];   // [qrow][krow] bf16 probabilities

    const int tid  = threadIdx.x;       // 0..127
    const int lane = tid & 31;
    const int wave = tid >> 5;          // 0..3, owns q rows [wave*16, wave*16+16)
    const int cn   = lane & 15;         // column index within a 16-wide tile
    const int half = lane >> 4;         // 0/1: lane half (wave32)
    const int h8   = half * 8;          // unused; A and B fragments both use hk
    const int hk   = half * 16;         // B-fragment K sub-offset

    const int qb = blockIdx.x;                         // query block 0..31
    const int bh = blockIdx.z * N_HEADS + blockIdx.y;  // flattened (b,h)
    const size_t base = (size_t)bh * SEQ * D_H;

    // ---------------- load Q tile -> LDS (fp32 -> bf16) ----------------
    {
        const float* Qg = Q + base + (size_t)qb * BR * D_H;
        #pragma unroll
        for (int i = 0; i < 8; ++i) {
            int idx4 = i * 128 + tid;          // 0..1023 float4 chunks
            int row  = idx4 >> 4;              // 64 rows
            int c4   = (idx4 & 15) << 2;       // column (multiple of 4)
            v4f qv = *(const v4f*)(Qg + row * D_H + c4);
            v4bf b; b.x = (__bf16)qv.x; b.y = (__bf16)qv.y;
                    b.z = (__bf16)qv.z; b.w = (__bf16)qv.w;
            *(v4bf*)&Qs[row * LDSTR + c4] = b;
        }
    }
    __syncthreads();

    // ------------- preload Q A-fragments (16x32 bf16 x 2) -------------
    // A layout: lane holds row M=lane%16; lanes0-15: K=0-15;
    // lanes16-31: K=16-31 (per 16x32 fragment, matches B fill).
    FragA aq[2];
    {
        const int arow = wave * 16 + cn;
        #pragma unroll
        for (int s = 0; s < 2; ++s) {
            aq[s].h[0] = *(const v8bf*)&Qs[arow * LDSTR + s * 32 + hk];
            aq[s].h[1] = *(const v8bf*)&Qs[arow * LDSTR + s * 32 + hk + 8];
        }
    }

    // ---------------- running state ----------------
    v8f o[4] = {};                    // O strip: 4 d-tiles, C-layout
    float m_run[8], l_run[8];
    #pragma unroll
    for (int v = 0; v < 8; ++v) { m_run[v] = -3.0e38f; l_run[v] = 0.0f; }

    const float sl2e = ATTN_SCALE * LOG2E;

    // ================= KV block loop =================
    for (int kb = 0; kb < NKB; ++kb) {
        __syncthreads();   // previous iteration's LDS reads complete

        const float* Kg = K + base + (size_t)kb * BC * D_H;
        const float* Vg = V + base + (size_t)kb * BC * D_H;

        // K tile -> Ks (row-major bf16)
        #pragma unroll
        for (int i = 0; i < 8; ++i) {
            int idx4 = i * 128 + tid;
            int row  = idx4 >> 4;
            int c4   = (idx4 & 15) << 2;
            v4f kv = *(const v4f*)(Kg + row * D_H + c4);
            v4bf b; b.x = (__bf16)kv.x; b.y = (__bf16)kv.y;
                    b.z = (__bf16)kv.z; b.w = (__bf16)kv.w;
            *(v4bf*)&Ks[row * LDSTR + c4] = b;
        }
        // V tile -> Vts transposed ([d][key])
        #pragma unroll
        for (int i = 0; i < 32; ++i) {
            int idx  = i * 128 + tid;          // coalesced global read
            int krow = idx >> 6;
            int d    = idx & 63;
            Vts[d * LDSTR + krow] = (__bf16)Vg[idx];
        }
        // prefetch next KV block into L2 (global_prefetch_b8); 128 lanes x 128B
        if (kb + 1 < NKB) {
            __builtin_prefetch(Kg + BC * D_H + tid * 32, 0, 3);
            __builtin_prefetch(Vg + BC * D_H + tid * 32, 0, 3);
        }
        __syncthreads();

        // ---------- S = Q * K^T  (16 x 64 strip per wave) ----------
        v8f s_[4];
        #pragma unroll
        for (int t = 0; t < 4; ++t) {
            // B fragment: column N = key (t*16+cn); K dim = d, contiguous in Ks row
            const __bf16* kp = &Ks[(t * 16 + cn) * LDSTR];
            FragA b0, b1;
            b0.h[0] = *(const v8bf*)(kp + hk);
            b0.h[1] = *(const v8bf*)(kp + hk + 8);
            b1.h[0] = *(const v8bf*)(kp + 32 + hk);
            b1.h[1] = *(const v8bf*)(kp + 32 + hk + 8);
            v8f acc = {};
            acc = __builtin_amdgcn_wmma_f32_16x16x32_bf16(
                      false, aq[0].v, false, b0.v, (short)0, acc, false, false);
            acc = __builtin_amdgcn_wmma_f32_16x16x32_bf16(
                      false, aq[1].v, false, b1.v, (short)0, acc, false, false);
            s_[t] = acc;
        }

        // ---------- online softmax (C-layout: VGPR v = row v+8*half) ----------
        #pragma unroll
        for (int v = 0; v < 8; ++v) {
            float mx = fmaxf(fmaxf(s_[0][v], s_[1][v]), fmaxf(s_[2][v], s_[3][v]));
            mx = row16_max(mx);                       // DPP, stays in 16-lane half
            float m_new = fmaxf(m_run[v], mx);
            float corr  = exp2f((m_run[v] - m_new) * sl2e);
            m_run[v] = m_new;
            l_run[v] *= corr;
            float rs = 0.0f;
            #pragma unroll
            for (int t = 0; t < 4; ++t) {
                float p = exp2f((s_[t][v] - m_new) * sl2e);
                s_[t][v] = p;
                rs += p;
                o[t][v] *= corr;                      // rescale O rows
            }
            l_run[v] += row16_sum(rs);                // DPP row sum
        }

        // ---------- P (bf16) -> LDS, per-wave disjoint 16-row strip ----------
        {
            const int prow = wave * 16 + half * 8;
            #pragma unroll
            for (int v = 0; v < 8; ++v)
                #pragma unroll
                for (int t = 0; t < 4; ++t)
                    Ps[(prow + v) * LDSTR + t * 16 + cn] = (__bf16)s_[t][v];
        }
        // P strip is per-wave private; same-wave LDS ops are in-order, so a
        // dscnt drain + compiler fence is enough (no workgroup barrier needed).
        asm volatile("s_wait_dscnt 0x0" ::: "memory");

        // ---------- O += P * V ----------
        FragA ap[2];
        {
            const int arow = wave * 16 + cn;
            #pragma unroll
            for (int s = 0; s < 2; ++s) {
                ap[s].h[0] = *(const v8bf*)&Ps[arow * LDSTR + s * 32 + hk];
                ap[s].h[1] = *(const v8bf*)&Ps[arow * LDSTR + s * 32 + hk + 8];
            }
        }
        #pragma unroll
        for (int t = 0; t < 4; ++t) {
            // B fragment: column N = d (t*16+cn); K dim = key, contiguous in Vts row
            const __bf16* vp = &Vts[(t * 16 + cn) * LDSTR];
            FragA b0, b1;
            b0.h[0] = *(const v8bf*)(vp + hk);
            b0.h[1] = *(const v8bf*)(vp + hk + 8);
            b1.h[0] = *(const v8bf*)(vp + 32 + hk);
            b1.h[1] = *(const v8bf*)(vp + 32 + hk + 8);
            o[t] = __builtin_amdgcn_wmma_f32_16x16x32_bf16(
                       false, ap[0].v, false, b0.v, (short)0, o[t], false, false);
            o[t] = __builtin_amdgcn_wmma_f32_16x16x32_bf16(
                       false, ap[1].v, false, b1.v, (short)0, o[t], false, false);
        }
    }

    // ---------------- epilogue: O / l -> global ----------------
    {
        float* Og = O + base + (size_t)qb * BR * D_H;
        const int rbase = wave * 16 + half * 8;
        #pragma unroll
        for (int v = 0; v < 8; ++v) {
            float inv = 1.0f / l_run[v];
            #pragma unroll
            for (int t = 0; t < 4; ++t)
                Og[(rbase + v) * D_H + t * 16 + cn] = o[t][v] * inv;
        }
    }
}

extern "C" void kernel_launch(void* const* d_in, const int* in_sizes, int n_in,
                              void* d_out, int out_size, void* d_ws, size_t ws_size,
                              hipStream_t stream) {
    (void)in_sizes; (void)n_in; (void)out_size; (void)d_ws; (void)ws_size;
    const float* q = (const float*)d_in[0];
    const float* k = (const float*)d_in[1];
    const float* v = (const float*)d_in[2];
    float* out = (float*)d_out;

    dim3 grid(SEQ / BR, N_HEADS, N_BATCH);   // (32, 16, 4) = 2048 workgroups
    dim3 block(128);                         // 4 wave32
    fa_fwd_bf16_wmma<<<grid, block, 0, stream>>>(q, k, v, out);
}
